// TimeSeriesAttention_53163105190620
// MI455X (gfx1250) — matrix operand order repaired and checked by reading, not compiled, not run on hardware
//
#include <hip/hip_runtime.h>
#include <hip/hip_bf16.h>
#include <math.h>
#include <stdint.h>

// ---------------------------------------------------------------------------
// Problem constants (reference: B=16, T=2048, D=512, F=256, kernel=1)
// ---------------------------------------------------------------------------
#define BATCH  16
#define SEQ    2048
#define DIM    512
#define FILT   256
#define ROWS   (BATCH * SEQ)        // 32768 flattened rows
#define QKVC   (3 * FILT)           // 768

// CDNA5 async global->LDS staging (ASYNCcnt-tracked, no VGPR round trip).
#define USE_ASYNC 1

typedef __attribute__((ext_vector_type(16))) __bf16 v16bf;
typedef __attribute__((ext_vector_type(8)))  __bf16 v8bf;
typedef __attribute__((ext_vector_type(8)))  float  v8f;

union AFrag { v16bf v; v8bf h[2]; };

// D = A(16x32 bf16) * B(32x16 bf16) + C(16x16 f32)
__device__ __forceinline__ v8f wmma_bf16(v16bf a, v16bf b, v8f c) {
    return __builtin_amdgcn_wmma_f32_16x16x32_bf16(
        /*neg_a=*/false, a, /*neg_b=*/false, b,
        /*c_mod=*/(short)0, c, /*reuse_a=*/false, /*reuse_b=*/false);
}

// A fragment (16x32, MxK), row-major source with element stride `ld`.
__device__ __forceinline__ v16bf load_afrag(const __bf16* base, int ld, int lane) {
    const int row = lane & 15, g = lane >> 4;
    AFrag f;
    f.h[0] = *(const v8bf*)(base + (size_t)row * ld + g * 8);
    f.h[1] = *(const v8bf*)(base + (size_t)row * ld + 16 + g * 8);
    return f.v;
}

// B fragment (32x16, KxN) from a column-major ("B-transposed") tile:
// src[n*ld + k] = B[k][n]; lane half g covers K=8g..8g+7 and 16+8g..16+8g+7.
__device__ __forceinline__ v16bf load_bfrag(const __bf16* base, int ld, int lane) {
    const int n = lane & 15, g = lane >> 4;
    AFrag f;
    f.h[0] = *(const v8bf*)(base + (size_t)n * ld + g * 8);
    f.h[1] = *(const v8bf*)(base + (size_t)n * ld + 16 + g * 8);
    return f.v;
}

// Stage one 16-byte chunk global -> LDS.
__device__ __forceinline__ void stage_b128(const __bf16* g, __bf16* l) {
#if USE_ASYNC
    // ISA 15.18.3: GLOBAL_LOAD_ASYNC_TO_LDS_B128 — VDST = LDS byte address VGPR,
    // VADDR = 64-bit global address. Flat LDS addresses truncate to LDS offset.
    asm volatile("global_load_async_to_lds_b128 %0, %1, off"
                 :: "v"((uint32_t)(uintptr_t)l), "v"(g)
                 : "memory");
#else
    *(v8bf*)l = *(const v8bf*)g;
#endif
}

__device__ __forceinline__ void stage_wait() {
#if USE_ASYNC
    asm volatile("s_wait_asynccnt 0x0" ::: "memory");
#endif
}

// ---------------------------------------------------------------------------
// fp32 -> bf16 convert (bandwidth trivial)
// ---------------------------------------------------------------------------
__global__ void f32_to_bf16_kernel(const float* __restrict__ src,
                                   __bf16* __restrict__ dst, int n) {
    int i = blockIdx.x * blockDim.x + threadIdx.x;
    if (i < n) dst[i] = (__bf16)src[i];
}

// fp32 [R][C] -> bf16 transposed [C][R]  (weights only: tiny, one shot)
__global__ void transpose_f32_to_bf16_kernel(const float* __restrict__ src,
                                             __bf16* __restrict__ dst, int R, int C) {
    int i = blockIdx.x * blockDim.x + threadIdx.x;
    if (i < R * C) {
        int r = i / C, c = i % C;
        dst[(size_t)c * R + r] = (__bf16)src[(size_t)r * C + c];
    }
}

// bf16 V slice of QKV[row][768] -> VT[b][f][t] (tiled 32x32 LDS transpose)
__global__ __launch_bounds__(256)
void vtranspose_kernel(const __bf16* __restrict__ qkv, __bf16* __restrict__ vt) {
    __shared__ __bf16 tile[32][33];
    const int b  = blockIdx.z;
    const int t0 = blockIdx.x * 32, f0 = blockIdx.y * 32;
    const int tx = threadIdx.x, ty = threadIdx.y;  // 32 x 8
#pragma unroll
    for (int i = 0; i < 4; ++i) {
        const int r = ty + 8 * i;  // t within tile
        tile[r][tx] = qkv[((size_t)b * SEQ + t0 + r) * QKVC + 2 * FILT + f0 + tx];
    }
    __syncthreads();
#pragma unroll
    for (int i = 0; i < 4; ++i) {
        const int r = ty + 8 * i;  // f within tile
        vt[((size_t)b * FILT + f0 + r) * SEQ + t0 + tx] = tile[tx][r];
    }
}

// ---------------------------------------------------------------------------
// bf16 WMMA GEMM with pre-transposed B:  C[M,N] = A[M,K] * BT[N,K]^T + bias
//   MODE 0: outb[m*ldc+n] = bf16(acc + bias[n])
//   MODE 1: outf[m*ldc+n] = acc + bias[n] + resid[m*ldc+n]
// Block = 64x128, 128 threads (4 waves); each wave computes 32x64 (8 WMMA/K-step).
// Templated on MODE so each epilogue is straight-line code.
// ---------------------------------------------------------------------------
template <int MODE>
__global__ __launch_bounds__(128)
void gemm_bf16_kernel(const __bf16* __restrict__ A, int lda,
                      const __bf16* __restrict__ BT,   // [N][K] row-major
                      int K, const float* __restrict__ bias,
                      __bf16* __restrict__ outb, float* __restrict__ outf,
                      const float* __restrict__ resid, int ldc) {
    __shared__ alignas(16) __bf16 Ash[64 * 40];    // 64 rows x 32 K (+8 pad)
    __shared__ alignas(16) __bf16 BTsh[128 * 40];  // 128 cols x 32 K (+8 pad)

    const int tid = threadIdx.x, lane = tid & 31, wave = tid >> 5;
    const int m0 = blockIdx.x * 64, n0 = blockIdx.y * 128;
    const int wr = (wave & 1) * 32, wc = (wave >> 1) * 64;

    v8f acc[2][4];
#pragma unroll
    for (int i = 0; i < 2; ++i)
#pragma unroll
        for (int j = 0; j < 4; ++j)
#pragma unroll
            for (int r = 0; r < 8; ++r) acc[i][j][r] = 0.0f;

    for (int k0 = 0; k0 < K; k0 += 32) {
        __syncthreads();
        // --- stage A tile (64x32): thread t -> row t/2, 16-elem half ---
        {
            const int r = tid >> 1, c = (tid & 1) * 16;
            const __bf16* src = A + (size_t)(m0 + r) * lda + k0 + c;
            stage_b128(src,     &Ash[r * 40 + c]);
            stage_b128(src + 8, &Ash[r * 40 + c + 8]);
        }
        // --- stage BT tile (128x32): thread t -> row t, 32 elems ---
        {
            const __bf16* src = BT + (size_t)(n0 + tid) * K + k0;
#pragma unroll
            for (int j = 0; j < 4; ++j)
                stage_b128(src + j * 8, &BTsh[tid * 40 + j * 8]);
        }
        stage_wait();
        __syncthreads();

        const v16bf a0 = load_afrag(&Ash[(wr)      * 40], 40, lane);
        const v16bf a1 = load_afrag(&Ash[(wr + 16) * 40], 40, lane);
#pragma unroll
        for (int j = 0; j < 4; ++j) {
            const v16bf b = load_bfrag(&BTsh[(wc + j * 16) * 40], 40, lane);
            acc[0][j] = wmma_bf16(a0, b, acc[0][j]);
            acc[1][j] = wmma_bf16(a1, b, acc[1][j]);
        }
    }

    // --- epilogue: C layout = lane half g holds rows r+8g, col = lane&15 ---
    const int g = lane >> 4, nl = lane & 15;
#pragma unroll
    for (int i = 0; i < 2; ++i) {
#pragma unroll
        for (int r = 0; r < 8; ++r) {
            const int m = m0 + wr + i * 16 + r + 8 * g;
            if (MODE == 0) {
                __bf16* orow = outb + (size_t)m * ldc + n0 + wc + nl;
#pragma unroll
                for (int j = 0; j < 4; ++j)
                    orow[j * 16] = (__bf16)(acc[i][j][r] + bias[n0 + wc + j * 16 + nl]);
            } else {
                float* orow = outf + (size_t)m * ldc + n0 + wc + nl;
                const float* rrow = resid + (size_t)m * ldc + n0 + wc + nl;
#pragma unroll
                for (int j = 0; j < 4; ++j)
                    orow[j * 16] = acc[i][j][r] + bias[n0 + wc + j * 16 + nl] + rrow[j * 16];
            }
        }
    }
}

// ---------------------------------------------------------------------------
// Flash attention. Q/K from QKV[row][768]; V from pre-transposed VT[b][f][t].
// Grid: (SEQ/128, BATCH); 256 threads = 8 waves; each wave owns 16 queries,
// streams keys in blocks of 32, keeps O (16x256 f32) + Q frags in registers.
// ---------------------------------------------------------------------------
__global__ __launch_bounds__(256)
void attn_kernel(const __bf16* __restrict__ QKV,
                 const __bf16* __restrict__ VT,
                 __bf16* __restrict__ ctx) {
    __shared__ alignas(16) __bf16 Ksh[32 * 264];    // K block row-major  (16,896 B)
    __shared__ alignas(16) __bf16 VTsh[256 * 40];   // V block transposed (20,480 B)
    __shared__ alignas(16) __bf16 Pst[8 * 16 * 40]; // per-wave P staging (10,240 B)

    const int tid = threadIdx.x, lane = tid & 31, wave = tid >> 5;
    const int g = lane >> 4, nl = lane & 15;
    const size_t rowbase = (size_t)blockIdx.y * SEQ;           // batch row offset
    const __bf16* vtb = VT + (size_t)blockIdx.y * FILT * SEQ;  // batch VT base
    const int qrow = blockIdx.x * 128 + wave * 16;             // query row in batch

    // Preload this wave's Q fragments (16 rows x 256 f), 8 A-frags.
    v16bf qf[8];
#pragma unroll
    for (int c = 0; c < 8; ++c) {
        AFrag f;
        const __bf16* base = QKV + (rowbase + qrow + nl) * QKVC + c * 32;
        f.h[0] = *(const v8bf*)(base + g * 8);
        f.h[1] = *(const v8bf*)(base + 16 + g * 8);
        qf[c] = f.v;
    }

    v8f o[16];
#pragma unroll
    for (int j = 0; j < 16; ++j)
#pragma unroll
        for (int r = 0; r < 8; ++r) o[j][r] = 0.0f;
    float mrun[8], lsum[8];
#pragma unroll
    for (int r = 0; r < 8; ++r) { mrun[r] = -1e30f; lsum[r] = 0.0f; }

    __bf16* mypst = Pst + wave * (16 * 40);
    const float scale = 0.0625f;   // 1/sqrt(256)

    for (int s0 = 0; s0 < SEQ; s0 += 32) {
        __syncthreads();
        // --- stage K tile (32 keys x 256 f): thread t -> row t/8, 32-elem chunk ---
        {
            const int r = tid >> 3, c0 = (tid & 7) * 32;
            const __bf16* ksrc = QKV + (rowbase + s0 + r) * QKVC + FILT + c0;
#pragma unroll
            for (int j = 0; j < 4; ++j)
                stage_b128(ksrc + j * 8, &Ksh[r * 264 + c0 + j * 8]);
            if (s0 + 32 < SEQ)
                __builtin_prefetch(QKV + (rowbase + s0 + 32 + r) * QKVC + FILT + c0, 0, 1);
        }
        // --- stage VT tile (256 f x 32 keys): thread t -> f row t ---
        {
            const __bf16* vsrc = vtb + (size_t)tid * SEQ + s0;
#pragma unroll
            for (int j = 0; j < 4; ++j)
                stage_b128(vsrc + j * 8, &VTsh[tid * 40 + j * 8]);
            if (s0 + 32 < SEQ)
                __builtin_prefetch(vsrc + 32, 0, 1);
        }
        stage_wait();
        __syncthreads();

        // --- S = Q * K^T : two 16x16 tiles over this 32-key block ---
        v8f st0, st1;
#pragma unroll
        for (int r = 0; r < 8; ++r) { st0[r] = 0.0f; st1[r] = 0.0f; }
#pragma unroll
        for (int c = 0; c < 8; ++c) {
            const v16bf kb0 = load_bfrag(Ksh + 0  * 264 + c * 32, 264, lane);
            const v16bf kb1 = load_bfrag(Ksh + 16 * 264 + c * 32, 264, lane);
            st0 = wmma_bf16(qf[c], kb0, st0);
            st1 = wmma_bf16(qf[c], kb1, st1);
        }

        // --- online softmax (rows live across 16 lanes of each half) ---
        float p0[8], p1[8], alpha[8];
#pragma unroll
        for (int r = 0; r < 8; ++r) {
            const float a = st0[r] * scale;
            const float b = st1[r] * scale;
            float mt = fmaxf(a, b);
            mt = fmaxf(mt, __shfl_xor(mt, 1, 32));
            mt = fmaxf(mt, __shfl_xor(mt, 2, 32));
            mt = fmaxf(mt, __shfl_xor(mt, 4, 32));
            mt = fmaxf(mt, __shfl_xor(mt, 8, 32));
            const float mnew = fmaxf(mrun[r], mt);
            alpha[r] = __expf(mrun[r] - mnew);
            mrun[r] = mnew;
            p0[r] = __expf(a - mnew);
            p1[r] = __expf(b - mnew);
            float rs = p0[r] + p1[r];
            rs += __shfl_xor(rs, 1, 32);
            rs += __shfl_xor(rs, 2, 32);
            rs += __shfl_xor(rs, 4, 32);
            rs += __shfl_xor(rs, 8, 32);
            lsum[r] = lsum[r] * alpha[r] + rs;
        }
#pragma unroll
        for (int j = 0; j < 16; ++j)
#pragma unroll
            for (int r = 0; r < 8; ++r) o[j][r] *= alpha[r];

        // --- C-layout P -> row-major LDS -> A-layout fragment ---
#pragma unroll
        for (int r = 0; r < 8; ++r) {
            mypst[(r + 8 * g) * 40 + nl]      = (__bf16)p0[r];
            mypst[(r + 8 * g) * 40 + 16 + nl] = (__bf16)p1[r];
        }
        AFrag pf;
        pf.h[0] = *(const v8bf*)(mypst + nl * 40 + g * 8);
        pf.h[1] = *(const v8bf*)(mypst + nl * 40 + 16 + g * 8);

        // --- O += P(16x32) * V(32x256): 16 N-tiles ---
#pragma unroll
        for (int j = 0; j < 16; ++j) {
            const v16bf vb = load_bfrag(VTsh + (j * 16) * 40, 40, lane);
            o[j] = wmma_bf16(pf.v, vb, o[j]);
        }
    }

    // --- epilogue: context = O / l, bf16 to workspace ---
#pragma unroll
    for (int r = 0; r < 8; ++r) {
        const int m = qrow + r + 8 * g;
        __bf16* crow = ctx + (rowbase + m) * FILT + nl;
        const float inv = 1.0f / lsum[r];
#pragma unroll
        for (int j = 0; j < 16; ++j)
            crow[j * 16] = (__bf16)(o[j][r] * inv);
    }
}

// ---------------------------------------------------------------------------
// Host launcher
// ---------------------------------------------------------------------------
extern "C" void kernel_launch(void* const* d_in, const int* in_sizes, int n_in,
                              void* d_out, int out_size, void* d_ws, size_t ws_size,
                              hipStream_t stream) {
    const float* X    = (const float*)d_in[0];  // [32768,512]
    const float* Wqkv = (const float*)d_in[1];  // [512,768]
    const float* bqkv = (const float*)d_in[2];  // [768]
    const float* Wout = (const float*)d_in[3];  // [256,512]
    const float* bout = (const float*)d_in[4];  // [512]
    float* out = (float*)d_out;

    // Workspace carve-up (256-byte aligned)
    char* ws = (char*)d_ws;
    size_t off = 0;
    auto carve = [&](size_t bytes) { char* p = ws + off; off += (bytes + 255) & ~(size_t)255; return p; };
    __bf16* Xb     = (__bf16*)carve((size_t)ROWS * DIM  * 2);   // 33.6 MB
    __bf16* WqkvT  = (__bf16*)carve((size_t)DIM  * QKVC * 2);   // [768][512]
    __bf16* WoutT  = (__bf16*)carve((size_t)FILT * DIM  * 2);   // [512][256]
    __bf16* QKVb   = (__bf16*)carve((size_t)ROWS * QKVC * 2);   // 50.3 MB
    __bf16* VTg    = (__bf16*)carve((size_t)ROWS * FILT * 2);   // [B][F][T], 16.8 MB
    __bf16* CTXb   = (__bf16*)carve((size_t)ROWS * FILT * 2);   // 16.8 MB
    (void)ws_size;

    // 1) converts: activations linear, weights transposed
    {
        int n = ROWS * DIM;
        f32_to_bf16_kernel<<<(n + 255) / 256, 256, 0, stream>>>(X, Xb, n);
        n = DIM * QKVC;
        transpose_f32_to_bf16_kernel<<<(n + 255) / 256, 256, 0, stream>>>(Wqkv, WqkvT, DIM, QKVC);
        n = FILT * DIM;
        transpose_f32_to_bf16_kernel<<<(n + 255) / 256, 256, 0, stream>>>(Wout, WoutT, FILT, DIM);
    }

    // 2) QKV projection: [32768,512] x [512,768] + b_qkv -> bf16 QKV buffer
    gemm_bf16_kernel<0><<<dim3(ROWS / 64, QKVC / 128), 128, 0, stream>>>(
        Xb, DIM, WqkvT, DIM, bqkv,
        QKVb, nullptr, nullptr, QKVC);

    // 3) V -> VT[b][f][t] one-shot transpose
    vtranspose_kernel<<<dim3(SEQ / 32, FILT / 32, BATCH), dim3(32, 8), 0, stream>>>(QKVb, VTg);

    // 4) Flash attention -> bf16 context
    attn_kernel<<<dim3(SEQ / 128, BATCH), 256, 0, stream>>>(QKVb, VTg, CTXb);

    // 5) Output projection + bias + residual -> fp32 d_out
    gemm_bf16_kernel<1><<<dim3(ROWS / 64, DIM / 128), 128, 0, stream>>>(
        CTXb, FILT, WoutT, FILT, bout,
        nullptr, out, X, DIM);
}
